// CustomMHA_50448685859328
// MI455X (gfx1250) — matrix operand order repaired and checked by reading, not compiled, not run on hardware
//
#include <hip/hip_runtime.h>

// ---------------------------------------------------------------------------
// CDNA5 (gfx1250) wave32 WMMA MHA forward.
// Pipeline: f32->f16 convert, QKV GEMM (wmma, 4x2 register blocking),
//           flash attention (wmma + TDM tensor_load_to_lds for V tiles),
//           output GEMM (wmma, f32 out).
// Workspace layout (48 MiB total):
//   [ 0MiB)  Xh     f16 [4096,1024]
//   [ 8MiB)  Wqkvh  f16 [3072,1024]
//   [14MiB)  Woh    f16 [1024,1024]
//   [16MiB)  QKVh   f16 [4096,3072]  (== [B,S,3,H,dh])
//   [40MiB)  AOh    f16 [4096,1024]
// ---------------------------------------------------------------------------

typedef __attribute__((ext_vector_type(16))) _Float16     v16h;
typedef __attribute__((ext_vector_type(8)))  _Float16     v8h;
typedef __attribute__((ext_vector_type(8)))  float        v8f;
typedef __attribute__((ext_vector_type(4)))  unsigned int v4u;
typedef __attribute__((ext_vector_type(8)))  int          v8i;
typedef __attribute__((ext_vector_type(4)))  int          v4i;

__device__ __forceinline__ v8f wmma_f16(v16h a, v16h b, v8f c) {
  return __builtin_amdgcn_wmma_f32_16x16x32_f16(false, a, false, b,
                                                (short)0, c, false, false);
}

// A-fragment (16x32 f16, MxK): lane group g=(lane>>4): row=lane&15,
// elements i -> K = 16g + i (contiguous).
__device__ __forceinline__ v16h load_a_frag(const _Float16* A, int lda,
                                            int row0, int k0) {
  int lane = threadIdx.x & 31;
  const _Float16* p = A + (size_t)(row0 + (lane & 15)) * lda + k0 +
                      ((lane >> 4) << 4);
  v8h lo = *(const v8h*)p;         // K = kb .. kb+7
  v8h hi = *(const v8h*)(p + 8);   // K = kb+8 .. kb+15
  v16h o;
#pragma unroll
  for (int i = 0; i < 8; ++i) { o[i] = lo[i]; o[i + 8] = hi[i]; }
  return o;
}

// B-fragment (32x16 f16, KxN) from row-major W[N,K] (so B(k,n)=W[n][k]):
// lane group g: col n = lane&15, elements i -> K = 16g + i (contiguous 32B).
__device__ __forceinline__ v16h load_b_frag(const _Float16* W, int ldw,
                                            int n0, int k0) {
  int lane = threadIdx.x & 31;
  const _Float16* p = W + (size_t)(n0 + (lane & 15)) * ldw + k0 +
                      ((lane >> 4) << 4);
  return *(const v16h*)p;  // 32B aligned by construction
}

// ---------------------------------------------------------------------------
__global__ void f32_to_f16_kernel(const float* __restrict__ in,
                                  _Float16* __restrict__ out, int n) {
  int i = blockIdx.x * blockDim.x + threadIdx.x;
  int stride = gridDim.x * blockDim.x;
  for (; i < n; i += stride) out[i] = (_Float16)in[i];
}

// C[M,N] = A[M,K] @ W[N,K]^T ; 64x32 output per wave (4x2 tiles of 16x16),
// 8 WMMAs per 6 fragment loads per K-step. 4 waves/block.
template <bool F16OUT>
__global__ void __launch_bounds__(128)
gemm_nt_wmma(const _Float16* __restrict__ A, const _Float16* __restrict__ W,
             void* __restrict__ Cv, int N, int K) {
  int gw = blockIdx.x * (blockDim.x >> 5) + (threadIdx.x >> 5);
  int nbt = N >> 5;  // 32-wide column blocks
  int bm = gw / nbt;
  int bn = gw % nbt;
  int lane = threadIdx.x & 31;
  int row0 = bm << 6;
  int col0 = bn << 5;

  v8f acc[4][2] = {};
  for (int k0 = 0; k0 < K; k0 += 32) {
    v16h a[4], b[2];
#pragma unroll
    for (int i = 0; i < 4; ++i) a[i] = load_a_frag(A, K, row0 + (i << 4), k0);
#pragma unroll
    for (int j = 0; j < 2; ++j) b[j] = load_b_frag(W, K, col0 + (j << 4), k0);
#pragma unroll
    for (int i = 0; i < 4; ++i)
#pragma unroll
      for (int j = 0; j < 2; ++j) acc[i][j] = wmma_f16(a[i], b[j], acc[i][j]);
  }
  // C/D layout: element r -> row (lane>>4)*8 + r, col lane&15.
  int pr = (lane >> 4) << 3;
  int pc = lane & 15;
#pragma unroll
  for (int i = 0; i < 4; ++i)
#pragma unroll
    for (int j = 0; j < 2; ++j) {
      int rbase = row0 + (i << 4) + pr;
      int cbase = col0 + (j << 4) + pc;
      if constexpr (F16OUT) {
        _Float16* C = (_Float16*)Cv;
#pragma unroll
        for (int r = 0; r < 8; ++r)
          C[(size_t)(rbase + r) * N + cbase] = (_Float16)acc[i][j][r];
      } else {
        float* C = (float*)Cv;
#pragma unroll
        for (int r = 0; r < 8; ++r)
          C[(size_t)(rbase + r) * N + cbase] = acc[i][j][r];
      }
    }
}

// ---------------------------------------------------------------------------
// Issue a TDM load of a 2D tile (tile_dim1 rows x tile_dim0 f16 elements,
// row stride `stride` in f16 units) from `gsrc` into LDS at `ldsoff` bytes.
// Descriptor layout per CDNA5 ISA.
__device__ __forceinline__ void tdm_load_tile_f16(const _Float16* gsrc,
                                                  unsigned ldsoff,
                                                  unsigned tile_d0,
                                                  unsigned tile_d1,
                                                  unsigned stride) {
  unsigned long long ga = (unsigned long long)(const void*)gsrc;
  v4u g0;
  g0[0] = 1u;                                   // count=1, user mode
  g0[1] = ldsoff;                               // lds_addr (bytes)
  g0[2] = (unsigned)(ga & 0xFFFFFFFFu);         // global_addr[31:0]
  g0[3] = ((unsigned)(ga >> 32) & 0x01FFFFFFu)  // global_addr[56:32]
          | 0x80000000u;                        // type=2 ("image")
  v8i g1;
  g1[0] = 0x00010000;                 // workgroup_mask=0, data_size=1 (2B)
  g1[1] = (int)(tile_d0 << 16);       // tensor_dim0[15:0] in [63:48]
  g1[2] = (int)(tile_d1 << 16);       // tensor_dim1[15:0] in [95:80]
  g1[3] = (int)(tile_d0 << 16);       // tile_dim0 in [127:112]
  g1[4] = (int)tile_d1;               // tile_dim1 in [143:128], tile_dim2=0
  g1[5] = (int)stride;                // tensor_dim0_stride[31:0]
  g1[6] = 0;                          // stride hi, tensor_dim1_stride lo
  g1[7] = 0;
  v4i gz4 = {};                       // groups 2/3: zero (2D tensor)
  v8i gz8 = {};                       // extra descriptor slot (6-arg form)
  __builtin_amdgcn_tensor_load_to_lds(g0, g1, gz4, gz4, gz8, 0);
}

// ---------------------------------------------------------------------------
// Flash attention: one wave per (b, h, 16-row q tile). KV processed in
// chunks of 32 columns; V tiles DMA'd into LDS by the Tensor Data Mover.
__global__ void __launch_bounds__(32)
attn_flash_wmma(const _Float16* __restrict__ qkvh,  // [B,S,3,H,dh] f16
                _Float16* __restrict__ aoh) {       // [B,S,H*dh]  f16
  constexpr int S = 2048, H = 16, DH = 64, LD = 3 * H * DH;  // LD = 3072

  __shared__ __align__(32) _Float16 Ps[16 * 32];  // P tile, row-major
  __shared__ __align__(32) _Float16 Vs[32 * 64];  // V chunk, row-major

  int bidx = blockIdx.x;
  int qt = bidx & (S / 16 - 1);    // 0..127
  int h  = (bidx >> 7) & (H - 1);  // 0..15
  int b  = bidx >> 11;             // 0..1
  int lane = threadIdx.x;

  const _Float16* Qb = qkvh + (size_t)b * S * LD + h * DH;
  const _Float16* Kb = Qb + H * DH;
  const _Float16* Vb = Qb + 2 * H * DH;

  // Wave-relative LDS byte address of Vs: low 32 bits of the generic pointer
  // (ISA §10.2: LDS generic address truncates to the LDS offset).
  unsigned vs_lds = (unsigned)(unsigned long long)(const void*)&Vs[0];

  int qbase = qt << 4;
  int prow = (lane >> 4) << 3;  // C-tile row base for this half-wave
  int pcol = lane & 15;         // C-tile column for this lane

  // Q fragments (dh = 64 -> two K=32 slabs), reused for all KV chunks.
  v16h qa0 = load_a_frag(Qb, LD, qbase, 0);
  v16h qa1 = load_a_frag(Qb, LD, qbase, 32);

  v8f O0 = {}, O1 = {}, O2 = {}, O3 = {};
  float m[8], l[8];
#pragma unroll
  for (int r = 0; r < 8; ++r) { m[r] = -__builtin_inff(); l[r] = 0.0f; }

  int nchunks = (qbase + 16 + 31) >> 5;
  for (int c = 0; c < nchunks; ++c) {
    int kvb = c << 5;

    // TDM: DMA V rows kvb..kvb+31 (64 f16 each, stride LD) into Vs.
    // TDM is unordered vs DS ops: drain prior-iteration LDS reads first.
    asm volatile("s_wait_dscnt 0x0" ::: "memory");
    tdm_load_tile_f16(Vb + (size_t)kvb * LD, vs_lds, DH, 32, LD);

    // Scores: two 16x16 tiles (kv cols kvb..+15 and +16..+31), K=64.
    v8f s0 = {}, s1 = {};
    s0 = wmma_f16(qa0, load_b_frag(Kb, LD, kvb, 0), s0);
    s1 = wmma_f16(qa0, load_b_frag(Kb, LD, kvb + 16, 0), s1);
    s0 = wmma_f16(qa1, load_b_frag(Kb, LD, kvb, 32), s0);
    s1 = wmma_f16(qa1, load_b_frag(Kb, LD, kvb + 16, 32), s1);

    // Scale 1/sqrt(64), causal mask (faithful: *mask then zero -> -inf).
    float mc[8];
#pragma unroll
    for (int r = 0; r < 8; ++r) {
      int row = qbase + prow + r;
      float a = s0[r] * 0.125f;
      a = (kvb + pcol <= row) ? a : 0.0f;
      if (a == 0.0f) a = -__builtin_inff();
      float bb = s1[r] * 0.125f;
      bb = (kvb + 16 + pcol <= row) ? bb : 0.0f;
      if (bb == 0.0f) bb = -__builtin_inff();
      s0[r] = a; s1[r] = bb;
      mc[r] = fmaxf(a, bb);
    }
    // Row max: reduce across the 16 lanes of each half-wave.
#pragma unroll
    for (int off = 1; off < 16; off <<= 1)
#pragma unroll
      for (int r = 0; r < 8; ++r)
        mc[r] = fmaxf(mc[r], __shfl_xor(mc[r], off, 32));

    float rs[8];
#pragma unroll
    for (int r = 0; r < 8; ++r) {
      float mn = fmaxf(m[r], mc[r]);
      float alpha = __expf(m[r] - mn);
      float p0 = __expf(s0[r] - mn);
      float p1 = __expf(s1[r] - mn);
      Ps[(prow + r) * 32 + pcol]      = (_Float16)p0;
      Ps[(prow + r) * 32 + pcol + 16] = (_Float16)p1;
      rs[r] = p0 + p1;
      m[r] = mn;
      l[r] *= alpha;
      O0[r] *= alpha; O1[r] *= alpha; O2[r] *= alpha; O3[r] *= alpha;
    }
#pragma unroll
    for (int off = 1; off < 16; off <<= 1)
#pragma unroll
      for (int r = 0; r < 8; ++r) rs[r] += __shfl_xor(rs[r], off, 32);
#pragma unroll
    for (int r = 0; r < 8; ++r) l[r] += rs[r];

    // A-fragment of P from LDS.
    v16h pa = load_a_frag(Ps, 32, 0, 0);

    // Wait for the TDM tile before reading V fragments.
    __builtin_amdgcn_s_wait_tensorcnt(0);
    asm volatile("" ::: "memory");

    // O += P(16x32) @ Vchunk(32x64): 4 N-tiles, single K=32 WMMA each.
    int kbv = (lane >> 4) << 4;
#pragma unroll
    for (int t = 0; t < 4; ++t) {
      int nc = (t << 4) + pcol;
      v16h vb;
#pragma unroll
      for (int i = 0; i < 16; ++i) vb[i] = Vs[(kbv + i) * DH + nc];
      if      (t == 0) O0 = wmma_f16(pa, vb, O0);
      else if (t == 1) O1 = wmma_f16(pa, vb, O1);
      else if (t == 2) O2 = wmma_f16(pa, vb, O2);
      else             O3 = wmma_f16(pa, vb, O3);
    }
  }

  // Normalize and store attention output (f16, [B,S,H*dh]).
#pragma unroll
  for (int r = 0; r < 8; ++r) {
    int row = qbase + prow + r;
    float inv = 1.0f / l[r];
    size_t base = ((size_t)b * S + row) * (H * DH) + h * DH + pcol;
    aoh[base + 0]  = (_Float16)(O0[r] * inv);
    aoh[base + 16] = (_Float16)(O1[r] * inv);
    aoh[base + 32] = (_Float16)(O2[r] * inv);
    aoh[base + 48] = (_Float16)(O3[r] * inv);
  }
}

// ---------------------------------------------------------------------------
extern "C" void kernel_launch(void* const* d_in, const int* in_sizes, int n_in,
                              void* d_out, int out_size, void* d_ws,
                              size_t ws_size, hipStream_t stream) {
  const float* x    = (const float*)d_in[0];  // [2,2048,1024]
  const float* qkvw = (const float*)d_in[1];  // [3072,1024]
  const float* wo   = (const float*)d_in[2];  // [1024,1024]
  float* out = (float*)d_out;                 // [2,2048,1024]

  char* ws = (char*)d_ws;
  const size_t MiB = 1024 * 1024;
  _Float16* Xh    = (_Float16*)(ws + 0 * MiB);
  _Float16* Wqkvh = (_Float16*)(ws + 8 * MiB);
  _Float16* Woh   = (_Float16*)(ws + 14 * MiB);
  _Float16* QKVh  = (_Float16*)(ws + 16 * MiB);
  _Float16* AOh   = (_Float16*)(ws + 40 * MiB);

  f32_to_f16_kernel<<<4096, 256, 0, stream>>>(x, Xh, 4096 * 1024);
  f32_to_f16_kernel<<<3072, 256, 0, stream>>>(qkvw, Wqkvh, 3072 * 1024);
  f32_to_f16_kernel<<<1024, 256, 0, stream>>>(wo, Woh, 1024 * 1024);

  {  // QKVh[4096,3072] = Xh[4096,1024] @ Wqkvh[3072,1024]^T
    int N = 3072, K = 1024;
    int waves = (4096 / 64) * (N / 32);  // 6144
    gemm_nt_wmma<true><<<waves / 4, 128, 0, stream>>>(Xh, Wqkvh, QKVh, N, K);
  }

  attn_flash_wmma<<<2 * 16 * 128, 32, 0, stream>>>(QKVh, AOh);

  {  // out[4096,1024] = AOh[4096,1024] @ Woh[1024,1024]^T  (f32 out)
    int N = 1024, K = 1024;
    int waves = (4096 / 64) * (N / 32);  // 2048
    gemm_nt_wmma<false><<<waves / 4, 128, 0, stream>>>(AOh, Woh, out, N, K);
  }

  (void)in_sizes; (void)n_in; (void)out_size; (void)ws_size;
}
